// HybridBlock_39986145526076
// MI455X (gfx1250) — matrix operand order repaired and checked by reading, not compiled, not run on hardware
//
#include <hip/hip_runtime.h>

// ---------------------------------------------------------------------------
// CDNA5 / gfx1250 WMMA GNN hybrid block, round 3.
//  - All dense GEMMs: v_wmma_f32_16x16x32_bf16 (bf16 A/B, fp32 accum).
//  - Weight matrices pre-packed to WMMA-fragment-major bf16 (32B/lane loads).
//  - feature1 tile streamed to LDS via Tensor Data Mover (tensor_load_to_lds,
//    TENSORcnt), fp32 in LDS, converted during A-fragment assembly.
//  - Scatter-adds: global_atomic_add_f32.
// ---------------------------------------------------------------------------

typedef __attribute__((ext_vector_type(16))) __bf16    v16bf;
typedef __attribute__((ext_vector_type(8)))  float     v8f;
typedef __attribute__((ext_vector_type(4)))  unsigned  uint32x4;
typedef __attribute__((ext_vector_type(8)))  int       int32x8;
typedef __attribute__((ext_vector_type(4)))  int       int32x4;

#define WMMA_BF16(A, B, C) \
    __builtin_amdgcn_wmma_f32_16x16x32_bf16(false, (A), false, (B), (short)0, (C), false, false)

#define H      128
#define MID    256
#define LN_EPS 1e-5f

// ---------------------------------------------------------------------------
// Fragment helpers
// ---------------------------------------------------------------------------

// A fragment (16x32 bf16) from LDS bf16 row-major [rows][ld].
__device__ __forceinline__ v16bf lds_fragA(const __bf16* s, int lane, int m0, int k0, int ld) {
    const int m  = m0 + (lane & 15);
    const int kb = k0 + ((lane & 16) ? 8 : 0);
    const __bf16* row = s + m * ld;
    v16bf a;
#pragma unroll
    for (int j = 0; j < 8; ++j) {
        a[j]     = row[kb + j];
        a[j + 8] = row[kb + 16 + j];
    }
    return a;
}

// A fragment from LDS fp32 row-major [rows][ld] (convert on the fly).
__device__ __forceinline__ v16bf lds_fragA_f32(const float* s, int lane, int m0, int k0, int ld) {
    const int m  = m0 + (lane & 15);
    const int kb = k0 + ((lane & 16) ? 8 : 0);
    const float* row = s + m * ld;
    v16bf a;
#pragma unroll
    for (int j = 0; j < 8; ++j) {
        a[j]     = (__bf16)row[kb + j];
        a[j + 8] = (__bf16)row[kb + 16 + j];
    }
    return a;
}

// B fragment from packed fragment-major bf16 weights: one 32B vector per lane.
// Packed layout: frag (kt,nt) at ((kt*nTiles+nt)*512 + lane*16 + j).
__device__ __forceinline__ v16bf pkd_fragB(const __bf16* __restrict__ P, int lane, int kt, int nt, int nTiles) {
    return *(const v16bf*)(P + (((size_t)(kt * nTiles + nt)) << 9) + lane * 16);
}

// ---------------------------------------------------------------------------
// Kernel: pack fp32 weight [K][N] into WMMA-fragment-major bf16.
// B-layout: lanes 0-15 hold K=kt*32+{0..7,16..23}, lanes 16-31 hold
// K=kt*32+{8..15,24..31} (j=(K&7)+((K&16)?8:0)); column = nt*16 + (lane&15).
// ---------------------------------------------------------------------------
__global__ void pack_weight_kernel(const float* __restrict__ src, __bf16* __restrict__ dst, int K, int N) {
    int i = blockIdx.x * 256 + threadIdx.x;
    if (i >= K * N) return;
    int k = i / N, n = i - k * N;
    int kt = k >> 5, kin = k & 31;
    int nt = n >> 4, nin = n & 15;
    int lane = ((kin & 8) ? 16 : 0) + nin;
    int j    = (kin & 7) + ((kin & 16) ? 8 : 0);
    dst[(((size_t)(kt * (N >> 4) + nt)) << 9) + lane * 16 + j] = (__bf16)src[i];
}

// ---------------------------------------------------------------------------
// Kernel: zero accumulators.
// ---------------------------------------------------------------------------
__global__ void zero_kernel(float* __restrict__ p, int n) {
    int i = blockIdx.x * 256 + threadIdx.x;
    if (i < n) p[i] = 0.0f;
}

// ---------------------------------------------------------------------------
// Edge kernel: f1/f2 MLPs fused with gather(x[src])*f and atomic scatter-add.
// One block = 64 edges, 8 waves, 64 KB LDS.
// feature1 tile arrives via the Tensor Data Mover (async, TENSORcnt).
// ---------------------------------------------------------------------------
__global__ __launch_bounds__(256) void edge_conv_kernel(
    const float* __restrict__ x,
    const float* __restrict__ feature1, const float* __restrict__ feature2,
    const int*   __restrict__ pei,            // [2,E]
    const __bf16* __restrict__ Pf1a, const float* __restrict__ bf1a,
    const __bf16* __restrict__ Pf1b, const float* __restrict__ bf1b,
    const __bf16* __restrict__ Pf2a, const float* __restrict__ bf2a,
    const __bf16* __restrict__ Pf2b, const float* __restrict__ bf2b,
    float* __restrict__ agg1, float* __restrict__ agg2, int E)
{
    __shared__ float  featf[64 * H];     // 32 KB: feature1 tile fp32 (TDM target);
                                         //        later [0..2047]   = feature2 tile fp32
                                         //              [2048..2175] = src/dst index cache
    __shared__ __bf16 hidb[64 * MID];    // 32 KB: hidden activations bf16

    int* s_src = (int*)(featf + 64 * 32);
    int* s_dst = s_src + 64;

    const int tid  = threadIdx.x;
    const int w    = tid >> 5;
    const int lane = tid & 31;
    const int eb   = blockIdx.x * 64;

    // ---- phase 0: TDM async load of feature1 tile [64 x 128] fp32 into LDS.
    if (w == 0) {
        const unsigned lds_base = (unsigned)(uintptr_t)featf;  // low 32b = LDS offset
        const unsigned long long ga =
            (unsigned long long)(uintptr_t)(feature1 + (size_t)eb * H);
        uint32x4 g0;
        g0[0] = 1u;                                        // count=1, no gather
        g0[1] = lds_base;                                  // lds_addr
        g0[2] = (unsigned)(ga & 0xFFFFFFFFu);              // global_addr[31:0]
        g0[3] = (unsigned)((ga >> 32) & 0x01FFFFFFu)       // global_addr[56:32]
              | (2u << 30);                                // type = 2 (image)
        const unsigned rows_left = (unsigned)(E - eb);
        int32x8 g1;
        g1[0] = (2 << 16);                                 // data_size = 4B
        g1[1] = (int)(128u << 16);                         // tensor_dim0 = 128 (lo16)
        g1[2] = (int)((rows_left & 0xFFFFu) << 16);        // dim0 hi=0 | tensor_dim1 lo16
        g1[3] = (int)(((rows_left >> 16) & 0xFFFFu)        // tensor_dim1 hi16
              | (128u << 16));                             // tile_dim0 = 128
        g1[4] = 64;                                        // tile_dim1 = 64, tile_dim2 = 0
        g1[5] = 128;                                       // tensor_dim0_stride = 128
        g1[6] = 0;
        g1[7] = 0;
        const int32x4 z4 = {0, 0, 0, 0};
        const int32x8 z8 = {0, 0, 0, 0, 0, 0, 0, 0};
        __builtin_amdgcn_tensor_load_to_lds(g0, g1, z4, z4, z8, 0);
        __builtin_amdgcn_s_wait_tensorcnt(0);
    }
    __syncthreads();

    // ---- phase 1: hidden1 = relu(feat1 @ Wf1a + bf1a)  [64,256]; 8 tiles/wave
#pragma unroll 1
    for (int t = w * 8; t < w * 8 + 8; ++t) {
        const int mt = t >> 4, nt = t & 15;
        v8f acc = {};
#pragma unroll
        for (int kt = 0; kt < 4; ++kt) {
            v16bf A = lds_fragA_f32(featf, lane, mt * 16, kt * 32, H);
            v16bf B = pkd_fragB(Pf1a, lane, kt, nt, 16);
            acc = WMMA_BF16(A, B, acc);
        }
        const float bias = bf1a[nt * 16 + (lane & 15)];
        const int mo = (lane & 16) ? 8 : 0;
#pragma unroll
        for (int r = 0; r < 8; ++r) {
            float v = fmaxf(acc[r] + bias, 0.0f);
            hidb[(mt * 16 + mo + r) * MID + nt * 16 + (lane & 15)] = (__bf16)v;
        }
    }
    __syncthreads();

    // ---- phase 2: featf now free -> cache indices + feature2 tile
    if (tid < 64) {
        int e = min(eb + tid, E - 1);
        s_src[tid] = pei[e];
        s_dst[tid] = pei[E + e];
    }
    for (int i = tid; i < 64 * 32; i += 256) {
        int r = i >> 5, c = i & 31;
        int e = min(eb + r, E - 1);
        featf[i] = feature2[e * 32 + c];
    }
    __syncthreads();

    // ---- phase 3: f1 = hidden1 @ Wf1b + bf1b ; scatter agg1 += f1 * x[src]
#pragma unroll 1
    for (int t = w * 4; t < w * 4 + 4; ++t) {
        const int mt = t >> 3, nt = t & 7;
        v8f acc = {};
#pragma unroll
        for (int kt = 0; kt < 8; ++kt) {
            v16bf A = lds_fragA(hidb, lane, mt * 16, kt * 32, MID);
            v16bf B = pkd_fragB(Pf1b, lane, kt, nt, 8);
            acc = WMMA_BF16(A, B, acc);
        }
        const int   c    = nt * 16 + (lane & 15);
        const float bias = bf1b[c];
        const int   mo   = (lane & 16) ? 8 : 0;
#pragma unroll
        for (int r = 0; r < 8; ++r) {
            const int m = mt * 16 + mo + r;
            if (eb + m < E) {
                float f = acc[r] + bias;
                float v = f * x[(size_t)s_src[m] * H + c];
                atomicAdd(&agg1[(size_t)s_dst[m] * H + c], v);
            }
        }
    }
    __syncthreads();

    // ---- phase 4: hidden2 = relu(feat2 @ Wf2a + bf2a)   (K = 32)
#pragma unroll 1
    for (int t = w * 8; t < w * 8 + 8; ++t) {
        const int mt = t >> 4, nt = t & 15;
        v16bf A = lds_fragA_f32(featf, lane, mt * 16, 0, 32);
        v16bf B = pkd_fragB(Pf2a, lane, 0, nt, 16);
        v8f acc = {};
        acc = WMMA_BF16(A, B, acc);
        const float bias = bf2a[nt * 16 + (lane & 15)];
        const int mo = (lane & 16) ? 8 : 0;
#pragma unroll
        for (int r = 0; r < 8; ++r) {
            float v = fmaxf(acc[r] + bias, 0.0f);
            hidb[(mt * 16 + mo + r) * MID + nt * 16 + (lane & 15)] = (__bf16)v;
        }
    }
    __syncthreads();

    // ---- phase 5: f2 = hidden2 @ Wf2b + bf2b ; scatter agg2 += f2 * x[src]
#pragma unroll 1
    for (int t = w * 4; t < w * 4 + 4; ++t) {
        const int mt = t >> 3, nt = t & 7;
        v8f acc = {};
#pragma unroll
        for (int kt = 0; kt < 8; ++kt) {
            v16bf A = lds_fragA(hidb, lane, mt * 16, kt * 32, MID);
            v16bf B = pkd_fragB(Pf2b, lane, kt, nt, 8);
            acc = WMMA_BF16(A, B, acc);
        }
        const int   c    = nt * 16 + (lane & 15);
        const float bias = bf2b[c];
        const int   mo   = (lane & 16) ? 8 : 0;
#pragma unroll
        for (int r = 0; r < 8; ++r) {
            const int m = mt * 16 + mo + r;
            if (eb + m < E) {
                float f = acc[r] + bias;
                float v = f * x[(size_t)s_src[m] * H + c];
                atomicAdd(&agg2[(size_t)s_dst[m] * H + c], v);
            }
        }
    }
}

// ---------------------------------------------------------------------------
// GINE scatter: agg3[d2[e]] += x[s2[e]] + edge_attr[e]
// ---------------------------------------------------------------------------
__global__ void gine_scatter_kernel(
    const float* __restrict__ x, const float* __restrict__ edge_attr,
    const int* __restrict__ ei, float* __restrict__ agg3, int E)
{
    unsigned idx = blockIdx.x * 256u + threadIdx.x;
    if (idx >= (unsigned)E * H) return;
    int e = idx >> 7, c = idx & (H - 1);
    int s = ei[e], d = ei[E + e];
    atomicAdd(&agg3[(size_t)d * H + c], x[(size_t)s * H + c] + edge_attr[idx]);
}

// ---------------------------------------------------------------------------
// Node kernel: GEMM stack.  One block = 64 nodes, 8 waves, 64 KB LDS.
// ---------------------------------------------------------------------------
__global__ __launch_bounds__(256) void node_kernel(
    const float* __restrict__ x,
    const float* __restrict__ agg1, const float* __restrict__ agg2, const float* __restrict__ agg3,
    const __bf16* __restrict__ P1_rel, const float* __restrict__ b1_rel, const __bf16* __restrict__ P1_root,
    const __bf16* __restrict__ P2_rel, const float* __restrict__ b2_rel, const __bf16* __restrict__ P2_root,
    const __bf16* __restrict__ Pc,    const float* __restrict__ bc,
    const float* __restrict__ eps_p,
    const __bf16* __restrict__ Pg1,   const float* __restrict__ bg1,
    const float* __restrict__ ln_g,  const float* __restrict__ ln_b,
    const __bf16* __restrict__ Pg2,   const float* __restrict__ bg2,
    float* __restrict__ out, int N)
{
    __shared__ __bf16 xb[64 * H];     // 16 KB : x tile bf16, persistent
    __shared__ __bf16 stage[64 * H];  // 16 KB : agg1 -> agg2 -> g -> t
    __shared__ float  big[64 * H];    // 32 KB : cat[64][256] bf16  /  gy fp32

    __bf16* catb = (__bf16*)big;

    const int tid  = threadIdx.x;
    const int w    = tid >> 5;
    const int lane = tid & 31;
    const int nb   = blockIdx.x * 64;
    const int mo   = (lane & 16) ? 8 : 0;
    const int nloc = lane & 15;

    for (int i = tid; i < 64 * H; i += 256) {
        int r = i >> 7, c = i & (H - 1);
        int row = min(nb + r, N - 1);
        xb[i]    = (__bf16)x[(size_t)row * H + c];
        stage[i] = (__bf16)agg1[(size_t)row * H + c];
    }
    __syncthreads();

    // h1 = relu(agg1 @ W1_rel + b1_rel + x @ W1_root) -> catb[:, 0:128]
#pragma unroll 1
    for (int t = w * 4; t < w * 4 + 4; ++t) {
        const int mt = t >> 3, nt = t & 7;
        v8f acc = {};
#pragma unroll
        for (int kt = 0; kt < 4; ++kt) {
            acc = WMMA_BF16(lds_fragA(stage, lane, mt * 16, kt * 32, H),
                            pkd_fragB(P1_rel, lane, kt, nt, 8), acc);
            acc = WMMA_BF16(lds_fragA(xb, lane, mt * 16, kt * 32, H),
                            pkd_fragB(P1_root, lane, kt, nt, 8), acc);
        }
        const float bias = b1_rel[nt * 16 + nloc];
#pragma unroll
        for (int r = 0; r < 8; ++r)
            catb[(mt * 16 + mo + r) * 256 + nt * 16 + nloc] = (__bf16)fmaxf(acc[r] + bias, 0.0f);
    }
    __syncthreads();

    // stage <- agg2, then h2 -> catb[:, 128:256]
    for (int i = tid; i < 64 * H; i += 256) {
        int r = i >> 7, c = i & (H - 1);
        int row = min(nb + r, N - 1);
        stage[i] = (__bf16)agg2[(size_t)row * H + c];
    }
    __syncthreads();
#pragma unroll 1
    for (int t = w * 4; t < w * 4 + 4; ++t) {
        const int mt = t >> 3, nt = t & 7;
        v8f acc = {};
#pragma unroll
        for (int kt = 0; kt < 4; ++kt) {
            acc = WMMA_BF16(lds_fragA(stage, lane, mt * 16, kt * 32, H),
                            pkd_fragB(P2_rel, lane, kt, nt, 8), acc);
            acc = WMMA_BF16(lds_fragA(xb, lane, mt * 16, kt * 32, H),
                            pkd_fragB(P2_root, lane, kt, nt, 8), acc);
        }
        const float bias = b2_rel[nt * 16 + nloc];
#pragma unroll
        for (int r = 0; r < 8; ++r)
            catb[(mt * 16 + mo + r) * 256 + 128 + nt * 16 + nloc] = (__bf16)fmaxf(acc[r] + bias, 0.0f);
    }
    __syncthreads();

    // h = relu(cat @ Wc + bc) -> registers (4 tiles/wave)
    v8f hacc[4];
#pragma unroll 1
    for (int i = 0; i < 4; ++i) {
        const int t = w * 4 + i;
        const int mt = t >> 3, nt = t & 7;
        v8f acc = {};
#pragma unroll
        for (int kt = 0; kt < 8; ++kt)
            acc = WMMA_BF16(lds_fragA(catb, lane, mt * 16, kt * 32, 256),
                            pkd_fragB(Pc, lane, kt, nt, 8), acc);
        const float bias = bc[nt * 16 + nloc];
#pragma unroll
        for (int r = 0; r < 8; ++r) acc[r] = fmaxf(acc[r] + bias, 0.0f);
        hacc[i] = acc;
    }
    __syncthreads();

    // stage <- g = (1+eps)*x + agg3  (bf16)
    const float onePlusEps = 1.0f + eps_p[0];
    for (int i = tid; i < 64 * H; i += 256) {
        int r = i >> 7, c = i & (H - 1);
        int row = min(nb + r, N - 1);
        stage[i] = (__bf16)(onePlusEps * x[(size_t)row * H + c] + agg3[(size_t)row * H + c]);
    }
    __syncthreads();

    // gy = g @ Wg1 + bg1 -> big (fp32)
#pragma unroll 1
    for (int t = w * 4; t < w * 4 + 4; ++t) {
        const int mt = t >> 3, nt = t & 7;
        v8f acc = {};
#pragma unroll
        for (int kt = 0; kt < 4; ++kt)
            acc = WMMA_BF16(lds_fragA(stage, lane, mt * 16, kt * 32, H),
                            pkd_fragB(Pg1, lane, kt, nt, 8), acc);
        const float bias = bg1[nt * 16 + nloc];
#pragma unroll
        for (int r = 0; r < 8; ++r)
            big[(mt * 16 + mo + r) * H + nt * 16 + nloc] = acc[r] + bias;
    }
    __syncthreads();

    // LayerNorm + ReLU per row -> stage (bf16 t-buffer)
    if (tid < 64) {
        const float* row = big + tid * H;
        float s = 0.0f, s2 = 0.0f;
#pragma unroll
        for (int c = 0; c < H; ++c) { float v = row[c]; s += v; s2 += v * v; }
        const float mean = s * (1.0f / H);
        const float var  = s2 * (1.0f / H) - mean * mean;
        const float inv  = rsqrtf(var + LN_EPS);
#pragma unroll
        for (int c = 0; c < H; ++c) {
            float v = (row[c] - mean) * inv * ln_g[c] + ln_b[c];
            stage[tid * H + c] = (__bf16)fmaxf(v, 0.0f);
        }
    }
    __syncthreads();

    // h3 = t @ Wg2 + bg2 accumulated into hacc; out = h + h3
#pragma unroll 1
    for (int i = 0; i < 4; ++i) {
        const int t = w * 4 + i;
        const int mt = t >> 3, nt = t & 7;
        v8f acc = hacc[i];
#pragma unroll
        for (int kt = 0; kt < 4; ++kt)
            acc = WMMA_BF16(lds_fragA(stage, lane, mt * 16, kt * 32, H),
                            pkd_fragB(Pg2, lane, kt, nt, 8), acc);
        const int   c    = nt * 16 + nloc;
        const float bias = bg2[c];
#pragma unroll
        for (int r = 0; r < 8; ++r) {
            const int row = nb + mt * 16 + mo + r;
            if (row < N) out[(size_t)row * H + c] = acc[r] + bias;
        }
    }
}

// ---------------------------------------------------------------------------
// Launcher
// ---------------------------------------------------------------------------
static inline size_t frag_elems(int K, int N) { return (size_t)(K / 32) * (N / 16) * 512; }

extern "C" void kernel_launch(void* const* d_in, const int* in_sizes, int n_in,
                              void* d_out, int out_size, void* d_ws, size_t ws_size,
                              hipStream_t stream)
{
    const float* x         = (const float*)d_in[0];
    const float* feature1  = (const float*)d_in[1];
    const float* feature2  = (const float*)d_in[2];
    const float* edge_attr = (const float*)d_in[3];
    const int*   pei       = (const int*)d_in[4];
    const int*   ei        = (const int*)d_in[5];
    const float* Wf1a = (const float*)d_in[6];  const float* bf1a = (const float*)d_in[7];
    const float* Wf1b = (const float*)d_in[8];  const float* bf1b = (const float*)d_in[9];
    const float* Wf2a = (const float*)d_in[10]; const float* bf2a = (const float*)d_in[11];
    const float* Wf2b = (const float*)d_in[12]; const float* bf2b = (const float*)d_in[13];
    const float* W1_rel = (const float*)d_in[14]; const float* b1_rel = (const float*)d_in[15];
    const float* W1_root = (const float*)d_in[16];
    const float* W2_rel = (const float*)d_in[17]; const float* b2_rel = (const float*)d_in[18];
    const float* W2_root = (const float*)d_in[19];
    const float* eps_p = (const float*)d_in[20];
    const float* Wg1 = (const float*)d_in[21]; const float* bg1 = (const float*)d_in[22];
    const float* ln_g = (const float*)d_in[23]; const float* ln_b = (const float*)d_in[24];
    const float* Wg2 = (const float*)d_in[25]; const float* bg2 = (const float*)d_in[26];
    const float* Wc  = (const float*)d_in[27]; const float* bc  = (const float*)d_in[28];

    const int N = in_sizes[0] / H;
    const int E = in_sizes[1] / H;

    float* agg1 = (float*)d_ws;
    float* agg2 = agg1 + (size_t)N * H;
    float* agg3 = agg2 + (size_t)N * H;

    // Packed bf16 weight region (32B-aligned: 3*N*H*4 bytes is 32B-multiple).
    __bf16* pw = (__bf16*)(agg3 + (size_t)N * H);
    __bf16* Pf1a   = pw;  pw += frag_elems(H,   MID);
    __bf16* Pf1b   = pw;  pw += frag_elems(MID, H);
    __bf16* Pf2a   = pw;  pw += frag_elems(32,  MID);
    __bf16* Pf2b   = pw;  pw += frag_elems(MID, H);
    __bf16* P1_rel = pw;  pw += frag_elems(H, H);
    __bf16* P1_root= pw;  pw += frag_elems(H, H);
    __bf16* P2_rel = pw;  pw += frag_elems(H, H);
    __bf16* P2_root= pw;  pw += frag_elems(H, H);
    __bf16* Pg1    = pw;  pw += frag_elems(H, H);
    __bf16* Pg2    = pw;  pw += frag_elems(H, H);
    __bf16* Pc     = pw;  pw += frag_elems(2 * H, H);

    // weight packing (tiny, sequential on stream)
    #define PACK(SRC, DST, K, NN) \
        pack_weight_kernel<<<((K) * (NN) + 255) / 256, 256, 0, stream>>>((SRC), (DST), (K), (NN))
    PACK(Wf1a, Pf1a, H, MID);
    PACK(Wf1b, Pf1b, MID, H);
    PACK(Wf2a, Pf2a, 32, MID);
    PACK(Wf2b, Pf2b, MID, H);
    PACK(W1_rel,  P1_rel,  H, H);
    PACK(W1_root, P1_root, H, H);
    PACK(W2_rel,  P2_rel,  H, H);
    PACK(W2_root, P2_root, H, H);
    PACK(Wg1, Pg1, H, H);
    PACK(Wg2, Pg2, H, H);
    PACK(Wc,  Pc,  2 * H, H);
    #undef PACK

    const int nz = 3 * N * H;
    zero_kernel<<<(nz + 255) / 256, 256, 0, stream>>>(agg1, nz);

    edge_conv_kernel<<<(E + 63) / 64, 256, 0, stream>>>(
        x, feature1, feature2, pei,
        Pf1a, bf1a, Pf1b, bf1b, Pf2a, bf2a, Pf2b, bf2b,
        agg1, agg2, E);

    const unsigned gsN = ((unsigned)E * H + 255u) / 256u;
    gine_scatter_kernel<<<gsN, 256, 0, stream>>>(x, edge_attr, ei, agg3, E);

    node_kernel<<<(N + 63) / 64, 256, 0, stream>>>(
        x, agg1, agg2, agg3,
        P1_rel, b1_rel, P1_root, P2_rel, b2_rel, P2_root,
        Pc, bc, eps_p, Pg1, bg1, ln_g, ln_b, Pg2, bg2,
        (float*)d_out, N);
}
